// FlexAttention_26044681683511
// MI455X (gfx1250) — matrix operand order repaired and checked by reading, not compiled, not run on hardware
//
#include <hip/hip_runtime.h>

typedef __bf16 bf16;
typedef bf16  v16bf __attribute__((ext_vector_type(16)));
typedef bf16  v8bf  __attribute__((ext_vector_type(8)));
typedef float f32x8 __attribute__((ext_vector_type(8)));
typedef int   v4i   __attribute__((ext_vector_type(4)));

union BF16x16 { v16bf v; v8bf h[2]; bf16 e[16]; };
union F32x8   { f32x8 v; float e[8]; };

// Probe for the gfx1250 async Global->LDS path (ASYNCcnt); fall back cleanly.
#if __has_builtin(__builtin_amdgcn_global_load_async_to_lds_b128) && \
    __has_builtin(__builtin_amdgcn_s_wait_asynccnt)
#define USE_ASYNC_LDS 1
#define GLOBAL_AS __attribute__((address_space(1)))
#define SHARED_AS __attribute__((address_space(3)))
#else
#define USE_ASYNC_LDS 0
#endif

// ---- WMMA helpers (CDNA5 16x16x32 bf16, f32 accum) ------------------------

__device__ __forceinline__ f32x8 wmma_bf16(v16bf a, v16bf b, f32x8 c) {
  // (neg_a, A, neg_b, B, c_mod, C, reuse_a, reuse_b)
  return __builtin_amdgcn_wmma_f32_16x16x32_bf16(false, a, false, b, (short)0, c,
                                                 false, false);
}

// A-matrix 16x32 bf16 (M x K), row-major source with leading dim ldk.
// Layout (ISA 7.12.2): lanes 0-15 = rows, VGPR0-3 hold K=0..7, VGPR4-7 K=16..23;
// lanes 16-31 hold K=8..15 and K=24..31 of the same rows.
__device__ __forceinline__ v16bf load_a16x32(const bf16* base, int ldk, int k0, int lane) {
  const int r = lane & 15, hi = lane >> 4;
  const bf16* p = base + (size_t)r * ldk + (k0 + hi * 8);
  BF16x16 u;
  u.h[0] = *(const v8bf*)(p);
  u.h[1] = *(const v8bf*)(p + 16);
  return u.v;
}

// B-matrix 32x16 bf16 (K x N) taken as rows of W (i.e. W^T view):
// lane n = column n, same K order as A: lanes 0-15 K=0..7,16..23; lanes 16-31 K=8..15,24..31.
__device__ __forceinline__ v16bf load_b32x16(const bf16* base, int ldk, int k0, int lane) {
  const int c = lane & 15, hi = lane >> 4;
  const bf16* p = base + (size_t)c * ldk + (k0 + hi * 8);
  BF16x16 u;
  u.h[0] = *(const v8bf*)(p);
  u.h[1] = *(const v8bf*)(p + 16);
  return u.v;
}

// ---- f32 -> bf16 conversion ------------------------------------------------

__global__ void f32_to_bf16(const float* __restrict__ src, bf16* __restrict__ dst, int n) {
  int i = blockIdx.x * 256 + threadIdx.x;
  if (i < n) dst[i] = (bf16)src[i];
}

// ---- GEMM: C[M,N] = A[M,K] (bf16) @ W[N,K]^T (bf16) + bias, f32 out --------
// 128x128 block tile, 8 waves: 4 (M) x 2 (N), each wave 32x64 = 2x4 WMMA tiles.

__global__ __launch_bounds__(256) void gemm_xwt_bf16(const bf16* __restrict__ A,
                                                     const bf16* __restrict__ W,
                                                     const float* __restrict__ bias,
                                                     float* __restrict__ C,
                                                     int M, int N, int K) {
  const int lane = threadIdx.x & 31;
  const int wave = threadIdx.x >> 5;
  const int wm = wave & 3;
  const int wn = wave >> 2;
  const int m0 = blockIdx.x * 128 + wm * 32;
  const int n0 = blockIdx.y * 128 + wn * 64;

  F32x8 acc[2][4] = {};

  for (int k0 = 0; k0 < K; k0 += 32) {
    if (k0 + 32 < K) {
      __builtin_prefetch(A + (size_t)(m0 + (lane & 15)) * K + k0 + 32, 0, 1);
      __builtin_prefetch(W + (size_t)(n0 + (lane & 15)) * K + k0 + 32, 0, 1);
    }
    v16bf a0 = load_a16x32(A + (size_t)m0 * K,        K, k0, lane);
    v16bf a1 = load_a16x32(A + (size_t)(m0 + 16) * K, K, k0, lane);
    v16bf b0 = load_b32x16(W + (size_t)(n0 +  0) * K, K, k0, lane);
    v16bf b1 = load_b32x16(W + (size_t)(n0 + 16) * K, K, k0, lane);
    v16bf b2 = load_b32x16(W + (size_t)(n0 + 32) * K, K, k0, lane);
    v16bf b3 = load_b32x16(W + (size_t)(n0 + 48) * K, K, k0, lane);
    acc[0][0].v = wmma_bf16(a0, b0, acc[0][0].v);
    acc[0][1].v = wmma_bf16(a0, b1, acc[0][1].v);
    acc[0][2].v = wmma_bf16(a0, b2, acc[0][2].v);
    acc[0][3].v = wmma_bf16(a0, b3, acc[0][3].v);
    acc[1][0].v = wmma_bf16(a1, b0, acc[1][0].v);
    acc[1][1].v = wmma_bf16(a1, b1, acc[1][1].v);
    acc[1][2].v = wmma_bf16(a1, b2, acc[1][2].v);
    acc[1][3].v = wmma_bf16(a1, b3, acc[1][3].v);
  }

  // C/D layout: VGPR r -> row r (lanes 0-15) / row r+8 (lanes 16-31), col = lane&15
  const int c = lane & 15, hi = lane >> 4;
  #pragma unroll
  for (int i = 0; i < 2; ++i)
    #pragma unroll
    for (int j = 0; j < 4; ++j)
      #pragma unroll
      for (int r = 0; r < 8; ++r) {
        int row = m0 + 16 * i + r + 8 * hi;
        int col = n0 + 16 * j + c;
        float v = acc[i][j].e[r];
        if (bias) v += bias[col];
        C[(size_t)row * N + col] = v;
      }
}

// ---- RoPE (interleaved pairs) on f32 input, bf16 output --------------------

__global__ void rope_bf16(const float* __restrict__ Xf, const float* __restrict__ cosT,
                          const float* __restrict__ sinT, bf16* __restrict__ Xb,
                          int total_pairs, int Nn) {
  int i = blockIdx.x * 256 + threadIdx.x;
  if (i >= total_pairs) return;
  int d2 = i & 31;           // HEAD_DIM/2 = 32 pairs
  int h  = (i >> 5) & 15;    // 16 heads
  int bn = i >> 9;           // (b*N + n)
  int n  = bn % Nn;
  size_t base = (size_t)bn * 1024 + h * 64 + 2 * d2;
  float q1 = Xf[base], q2 = Xf[base + 1];
  float cs = cosT[n * 32 + d2], sn = sinT[n * 32 + d2];
  Xb[base]     = (bf16)(q1 * cs - q2 * sn);
  Xb[base + 1] = (bf16)(q1 * sn + q2 * cs);
}

// ---- Flash attention: per-wave 16 q-rows, 32-key steps ---------------------
// block = 128 threads (4 waves, same (b,h), 64 consecutive q rows).

__global__ __launch_bounds__(128) void attn_fwd(const bf16* __restrict__ Qb,
                                                const bf16* __restrict__ Kb,
                                                const bf16* __restrict__ Vb,
                                                float* __restrict__ Ctx,
                                                int Nn) {
  __shared__ bf16 Vt[32 * 64];       // 32 keys x 64 dims, shared by block
  __shared__ bf16 Pt[4][16 * 32];    // per-wave P staging (C-layout -> A-layout)

  const int tid  = threadIdx.x;
  const int lane = tid & 31;
  const int wave = tid >> 5;
  const int h = blockIdx.y;
  const int b = blockIdx.z;
  const int q0 = blockIdx.x * 64 + wave * 16;
  const int ld = 1024;  // row stride of [B,N,H*D]

  const bf16* Qh = Qb + ((size_t)b * Nn) * ld + h * 64;
  const bf16* Kh = Kb + ((size_t)b * Nn) * ld + h * 64;
  const bf16* Vh = Vb + ((size_t)b * Nn) * ld + h * 64;

  const int hi = lane >> 4, cidx = lane & 15;
  const int vkey = tid >> 2;          // 0..31
  const int vd   = (tid & 3) * 16;    // 0,16,32,48

  // Q tile (16x64) is loop-invariant: keep it in 16 VGPRs.
  v16bf qa0 = load_a16x32(Qh + (size_t)q0 * ld, ld, 0,  lane);
  v16bf qa1 = load_a16x32(Qh + (size_t)q0 * ld, ld, 32, lane);

  F32x8 o[4] = {};
  float mrow[8], lrow[8];
  #pragma unroll
  for (int r = 0; r < 8; ++r) { mrow[r] = -3.0e38f; lrow[r] = 0.0f; }

  for (int key0 = 0; key0 < Nn; key0 += 32) {
    // V tile (32 keys x 64 dims) staged to LDS; async copy overlaps the
    // QK^T WMMAs below, completion enforced by s_wait_asynccnt before the
    // barrier (each wave waits on its own ASYNCcnt; the barrier publishes).
    {
      const bf16* src = Vh + (size_t)(key0 + vkey) * ld + vd;
#if USE_ASYNC_LDS
      __builtin_amdgcn_global_load_async_to_lds_b128(
          (GLOBAL_AS v4i*)(void*)src,
          (SHARED_AS v4i*)(void*)&Vt[vkey * 64 + vd], 0, 0);
      __builtin_amdgcn_global_load_async_to_lds_b128(
          (GLOBAL_AS v4i*)(void*)(src + 8),
          (SHARED_AS v4i*)(void*)&Vt[vkey * 64 + vd + 8], 0, 0);
#else
      *(v8bf*)&Vt[vkey * 64 + vd]     = *(const v8bf*)src;
      *(v8bf*)&Vt[vkey * 64 + vd + 8] = *(const v8bf*)(src + 8);
#endif
    }

    // S = Q(16x64) @ K^T -> two 16x16 tiles, K-dim 64 = 2 WMMA each
    F32x8 s0 = {}, s1 = {};
    {
      v16bf bk0 = load_b32x16(Kh + (size_t)(key0)      * ld, ld, 0, lane);
      v16bf bk1 = load_b32x16(Kh + (size_t)(key0 + 16) * ld, ld, 0, lane);
      s0.v = wmma_bf16(qa0, bk0, s0.v);
      s1.v = wmma_bf16(qa0, bk1, s1.v);
      v16bf bk2 = load_b32x16(Kh + (size_t)(key0)      * ld, ld, 32, lane);
      v16bf bk3 = load_b32x16(Kh + (size_t)(key0 + 16) * ld, ld, 32, lane);
      s0.v = wmma_bf16(qa1, bk2, s0.v);
      s1.v = wmma_bf16(qa1, bk3, s1.v);
    }

    // online softmax (rows live across 16-lane groups in one VGPR)
    #pragma unroll
    for (int r = 0; r < 8; ++r) {
      float x0 = s0.e[r] * 0.125f;   // 1/sqrt(64)
      float x1 = s1.e[r] * 0.125f;
      float rmax = fmaxf(x0, x1);
      rmax = fmaxf(rmax, __shfl_xor(rmax, 1, 32));
      rmax = fmaxf(rmax, __shfl_xor(rmax, 2, 32));
      rmax = fmaxf(rmax, __shfl_xor(rmax, 4, 32));
      rmax = fmaxf(rmax, __shfl_xor(rmax, 8, 32));
      float mnew  = fmaxf(mrow[r], rmax);
      float alpha = __expf(mrow[r] - mnew);
      float p0 = __expf(x0 - mnew);
      float p1 = __expf(x1 - mnew);
      float rsum = p0 + p1;
      rsum += __shfl_xor(rsum, 1, 32);
      rsum += __shfl_xor(rsum, 2, 32);
      rsum += __shfl_xor(rsum, 4, 32);
      rsum += __shfl_xor(rsum, 8, 32);
      lrow[r] = lrow[r] * alpha + rsum;
      mrow[r] = mnew;
      #pragma unroll
      for (int d = 0; d < 4; ++d) o[d].e[r] *= alpha;
      Pt[wave][(r + 8 * hi) * 32 + cidx]      = (bf16)p0;
      Pt[wave][(r + 8 * hi) * 32 + 16 + cidx] = (bf16)p1;
    }

#if USE_ASYNC_LDS
    __builtin_amdgcn_s_wait_asynccnt(0);
#endif
    __syncthreads();

    // O += P(16x32) @ V(32x64): P as A-frag from LDS, V as 4 B-frags from LDS
    v16bf pa = load_a16x32(&Pt[wave][0], 32, 0, lane);
    const int kb = hi * 8;
    #pragma unroll
    for (int d = 0; d < 4; ++d) {
      BF16x16 u;
      #pragma unroll
      for (int vv = 0; vv < 8; ++vv) {
        u.e[2 * vv]     = Vt[(kb + 2 * vv + (vv >= 4 ? 8 : 0))     * 64 + d * 16 + cidx];
        u.e[2 * vv + 1] = Vt[(kb + 2 * vv + 1 + (vv >= 4 ? 8 : 0)) * 64 + d * 16 + cidx];
      }
      o[d].v = wmma_bf16(pa, u.v, o[d].v);
    }
    __syncthreads();
  }

  // epilogue: Ctx[b, q0+row, h*64 + d*16 + col] = O / l
  #pragma unroll
  for (int d = 0; d < 4; ++d)
    #pragma unroll
    for (int r = 0; r < 8; ++r) {
      int row = q0 + r + 8 * hi;
      float val = o[d].e[r] / lrow[r];
      Ctx[((size_t)(b * Nn + row)) * 1024 + h * 64 + d * 16 + cidx] = val;
    }
}

// ---- host orchestration ----------------------------------------------------

extern "C" void kernel_launch(void* const* d_in, const int* in_sizes, int n_in,
                              void* d_out, int out_size, void* d_ws, size_t ws_size,
                              hipStream_t stream) {
  const float* x  = (const float*)d_in[0];
  const float* fc = (const float*)d_in[1];
  const float* fs = (const float*)d_in[2];
  const float* Wq = (const float*)d_in[3];
  const float* Wk = (const float*)d_in[4];
  const float* Wv = (const float*)d_in[5];
  const float* Wo = (const float*)d_in[6];
  const float* bo = (const float*)d_in[7];
  float* out = (float*)d_out;

  const int DIMc = 1024;
  const int BN = in_sizes[0] / DIMc;   // B*N = 4096
  const int Nn = in_sizes[1] / 32;     // 2048
  const int B  = BN / Nn;              // 2

  char* ws = (char*)d_ws;
  size_t off = 0;
  bf16* xb  = (bf16*)(ws + off); off += (size_t)BN * DIMc * 2;
  bf16* wqb = (bf16*)(ws + off); off += (size_t)DIMc * DIMc * 2;
  bf16* wkb = (bf16*)(ws + off); off += (size_t)DIMc * DIMc * 2;
  bf16* wvb = (bf16*)(ws + off); off += (size_t)DIMc * DIMc * 2;
  bf16* wob = (bf16*)(ws + off); off += (size_t)DIMc * DIMc * 2;
  float* Qf = (float*)(ws + off); off += (size_t)BN * DIMc * 4;
  float* Kf = (float*)(ws + off); off += (size_t)BN * DIMc * 4;
  float* Vf = (float*)(ws + off); off += (size_t)BN * DIMc * 4;
  bf16* Qb  = (bf16*)(ws + off); off += (size_t)BN * DIMc * 2;
  bf16* Kb  = (bf16*)(ws + off); off += (size_t)BN * DIMc * 2;
  bf16* Vb  = (bf16*)(ws + off); off += (size_t)BN * DIMc * 2;
  float* ctxf = Qf;          // Qf dead after rope -> reuse
  bf16*  ctxb = (bf16*)Kf;   // Kf dead after rope -> reuse

  int nx = BN * DIMc;
  int nw = DIMc * DIMc;
  f32_to_bf16<<<(nx + 255) / 256, 256, 0, stream>>>(x,  xb,  nx);
  f32_to_bf16<<<(nw + 255) / 256, 256, 0, stream>>>(Wq, wqb, nw);
  f32_to_bf16<<<(nw + 255) / 256, 256, 0, stream>>>(Wk, wkb, nw);
  f32_to_bf16<<<(nw + 255) / 256, 256, 0, stream>>>(Wv, wvb, nw);
  f32_to_bf16<<<(nw + 255) / 256, 256, 0, stream>>>(Wo, wob, nw);

  dim3 ggrid(BN / 128, DIMc / 128);
  gemm_xwt_bf16<<<ggrid, 256, 0, stream>>>(xb, wqb, nullptr, Qf, BN, DIMc, DIMc);
  gemm_xwt_bf16<<<ggrid, 256, 0, stream>>>(xb, wkb, nullptr, Kf, BN, DIMc, DIMc);
  gemm_xwt_bf16<<<ggrid, 256, 0, stream>>>(xb, wvb, nullptr, Vf, BN, DIMc, DIMc);

  int pairs = BN * DIMc / 2;
  rope_bf16<<<(pairs + 255) / 256, 256, 0, stream>>>(Qf, fc, fs, Qb, pairs, Nn);
  rope_bf16<<<(pairs + 255) / 256, 256, 0, stream>>>(Kf, fc, fs, Kb, pairs, Nn);
  f32_to_bf16<<<(nx + 255) / 256, 256, 0, stream>>>(Vf, Vb, nx);

  dim3 agrid(Nn / 64, 16, B);
  attn_fwd<<<agrid, 128, 0, stream>>>(Qb, Kb, Vb, ctxf, Nn);

  f32_to_bf16<<<(nx + 255) / 256, 256, 0, stream>>>(ctxf, ctxb, nx);
  gemm_xwt_bf16<<<ggrid, 256, 0, stream>>>(ctxb, wob, bo, out, BN, DIMc, DIMc);
}
